// NonLocal_77790447665590
// MI455X (gfx1250) — compile-verified, matrix operand order checked by reading, not run on hardware
//
#include <hip/hip_runtime.h>

// Problem constants (from reference)
#define B_  8
#define C_  512
#define D_  256
#define T_  2048
#define EPS_ 1e-5f

// GEMM tiling: block tile 128x128, K-step 32, 256 threads = 8 waves (wave32),
// waves arranged 4(M) x 2(N), each wave owns 32x64 = 2x4 subtiles of 16x16.
// Double-buffered LDS, two K-steps per loop iteration (all K dims % 64 == 0).
#define BK 32
#define BM 128
#define BN 128

typedef __attribute__((ext_vector_type(16))) __bf16 bf16x16;
typedef __attribute__((ext_vector_type(8)))  __bf16 bf16x8;
typedef __attribute__((ext_vector_type(8)))  float  f32x8;

// ---------------------------------------------------------------------------
// WMMA helpers
// ---------------------------------------------------------------------------
__device__ __forceinline__ f32x8 wmma_bf16(bf16x16 a, bf16x16 b, f32x8 c) {
  // D = A(16x32 bf16) * B(32x16 bf16) + C(16x16 f32)
  return __builtin_amdgcn_wmma_f32_16x16x32_bf16(
      /*neg_a=*/false, a, /*neg_b=*/false, b,
      /*c_mod=*/(short)0, c, /*reuse_a=*/false, /*reuse_b=*/false);
}

// LDS tiles are stored [row][k] (A) and [col][k] (B), k contiguous, stride BK.
// Per ISA 16-bit A layout: lanes 0-15 hold K {half*8+0..7, 16+half*8+0..7}
// with half = lane>>4 -> two contiguous 16B runs per lane (ds_load_b128 x2).
__device__ __forceinline__ bf16x16 frag_from_lds(const __bf16* base, int row, int half) {
  const __bf16* p = base + row * BK + half * 8;
  bf16x8 lo = *(const bf16x8*)(p);        // K = half*8 .. +7
  bf16x8 hi = *(const bf16x8*)(p + 16);   // K = 16 + half*8 .. +7
  return __builtin_shufflevector(lo, hi, 0,1,2,3,4,5,6,7,8,9,10,11,12,13,14,15);
}

// Copy 128(rows) x 32(k) bf16 tile; global is row-major [row][k], k contiguous.
__device__ __forceinline__ void lds_copy_direct(__bf16* dst, const __bf16* src,
                                                int ld, int row0, int k0, int tid) {
  int r  = tid >> 1;            // 0..127
  int kk = (tid & 1) * 16;      // 0 or 16
  const __bf16* s = src + (size_t)(row0 + r) * ld + k0 + kk;
  *(bf16x8*)(dst + r * BK + kk)     = *(const bf16x8*)(s);
  *(bf16x8*)(dst + r * BK + kk + 8) = *(const bf16x8*)(s + 8);
}
__device__ __forceinline__ void pf_direct(const __bf16* src, int ld, int row0,
                                          int k0, int tid) {
  int r  = tid >> 1;
  int kk = (tid & 1) * 16;
  __builtin_prefetch(src + (size_t)(row0 + r) * ld + k0 + kk, 0, 3);
}

// Transpose-copy: global [k][n] (n contiguous) -> LDS [n][k]; 32k x 128n.
__device__ __forceinline__ void lds_copy_trans(__bf16* dst, const __bf16* src,
                                               int ld, int k0, int n0, int tid) {
  int ki = tid >> 3;            // 0..31
  int nb = (tid & 7) * 16;      // 0..112
  const __bf16* s = src + (size_t)(k0 + ki) * ld + n0 + nb;
  #pragma unroll
  for (int j = 0; j < 16; ++j) dst[(nb + j) * BK + ki] = s[j];
}
__device__ __forceinline__ void pf_trans(const __bf16* src, int ld, int k0,
                                         int n0, int tid) {
  int ki = tid >> 3;
  int nb = (tid & 7) * 16;
  __builtin_prefetch(src + (size_t)(k0 + ki) * ld + n0 + nb, 0, 3);
}

// Exp-convert transpose copy for attention B tile: f32 scores -> bf16 E in LDS.
__device__ __forceinline__ void lds_copy_exp(__bf16* dst, const float* Sb,
                                             const float* Mb, int k0, int n0, int tid) {
  int ki = tid >> 3;
  int nb = (tid & 7) * 16;
  const float* s = Sb + (size_t)(k0 + ki) * T_ + n0 + nb;
  float mr = Mb[k0 + ki];
  #pragma unroll
  for (int j = 0; j < 16; ++j) dst[(nb + j) * BK + ki] = (__bf16)__expf(s[j] - mr);
}
__device__ __forceinline__ void pf_exp(const float* Sb, int k0, int n0, int tid) {
  int ki = tid >> 3;
  int nb = (tid & 7) * 16;
  __builtin_prefetch(Sb + (size_t)(k0 + ki) * T_ + n0 + nb, 0, 3);
}

#define GEMM_PROLOGUE()                                   \
  int tid  = threadIdx.x;                                 \
  int wave = tid >> 5, lane = tid & 31;                   \
  int wm   = (wave >> 1) * 32, wn = (wave & 1) * 64;      \
  int col  = lane & 15, half = lane >> 4;                 \
  f32x8 zero8 = {0.f,0.f,0.f,0.f,0.f,0.f,0.f,0.f};        \
  f32x8 acc[2][4];                                        \
  _Pragma("unroll") for (int i = 0; i < 2; ++i)           \
  _Pragma("unroll") for (int j = 0; j < 4; ++j) acc[i][j] = zero8;

#define GEMM_MAC(la, lb)                                                  \
  {                                                                       \
    bf16x16 af[2], bfr[4];                                                \
    _Pragma("unroll") for (int i = 0; i < 2; ++i)                         \
      af[i] = frag_from_lds((la), wm + i * 16 + col, half);               \
    _Pragma("unroll") for (int j = 0; j < 4; ++j)                         \
      bfr[j] = frag_from_lds((lb), wn + j * 16 + col, half);              \
    _Pragma("unroll") for (int i = 0; i < 2; ++i)                         \
      _Pragma("unroll") for (int j = 0; j < 4; ++j)                       \
        acc[i][j] = wmma_bf16(af[i], bfr[j], acc[i][j]);                  \
  }

// ---------------------------------------------------------------------------
// Kernel 1: fused projections  Y[3D x T] = Wcat[3D x C] * X[C x T]  (per batch)
// Output layout: TPG[head][b][d][t] bf16, head 0=theta,1=phi,2=g, with bias.
// ---------------------------------------------------------------------------
__global__ __launch_bounds__(256)
void k_proj(const __bf16* __restrict__ Wcat, const __bf16* __restrict__ X,
            const float* __restrict__ Bcat, __bf16* __restrict__ TPG) {
  __shared__ __align__(16) __bf16 lds_a[2][BM * BK];
  __shared__ __align__(16) __bf16 lds_b[2][BN * BK];
  int b  = blockIdx.z;
  int m0 = blockIdx.y * BM;   // over 3D = 768
  int n0 = blockIdx.x * BN;   // over T
  const __bf16* Xb = X + (size_t)b * C_ * T_;
  GEMM_PROLOGUE();

  lds_copy_direct(lds_a[0], Wcat, C_, m0, 0, tid);
  lds_copy_trans (lds_b[0], Xb,  T_, 0, n0, tid);
  for (int k0 = 0; k0 < C_; k0 += 2 * BK) {
    __syncthreads();
    lds_copy_direct(lds_a[1], Wcat, C_, m0, k0 + BK, tid);
    lds_copy_trans (lds_b[1], Xb,  T_, k0 + BK, n0, tid);
    if (k0 + 2 * BK < C_) {
      pf_direct(Wcat, C_, m0, k0 + 2 * BK, tid);
      pf_trans (Xb,  T_, k0 + 2 * BK, n0, tid);
    }
    GEMM_MAC(lds_a[0], lds_b[0]);
    __syncthreads();
    if (k0 + 2 * BK < C_) {
      lds_copy_direct(lds_a[0], Wcat, C_, m0, k0 + 2 * BK, tid);
      lds_copy_trans (lds_b[0], Xb,  T_, k0 + 2 * BK, n0, tid);
      pf_direct(Wcat, C_, m0, k0 + 3 * BK, tid);
      pf_trans (Xb,  T_, k0 + 3 * BK, n0, tid);
    }
    GEMM_MAC(lds_a[1], lds_b[1]);
  }

  #pragma unroll
  for (int i = 0; i < 2; ++i)
    #pragma unroll
    for (int j = 0; j < 4; ++j)
      #pragma unroll
      for (int e = 0; e < 8; ++e) {
        int m = m0 + wm + i * 16 + e + 8 * half;       // 0..767
        int n = n0 + wn + j * 16 + col;
        int head = m >> 8, dch = m & (D_ - 1);
        TPG[((size_t)head * B_ + b) * D_ * T_ + (size_t)dch * T_ + n] =
            (__bf16)(acc[i][j][e] + Bcat[m]);
      }
}

// ---------------------------------------------------------------------------
// Kernel 2: scores  S[t][s] = sum_d theta[d][t] * phi[d][s]  (per batch, f32)
// ---------------------------------------------------------------------------
__global__ __launch_bounds__(256)
void k_scores(const __bf16* __restrict__ TPG, float* __restrict__ S) {
  __shared__ __align__(16) __bf16 lds_a[2][BM * BK];
  __shared__ __align__(16) __bf16 lds_b[2][BN * BK];
  int b  = blockIdx.z;
  int t0 = blockIdx.y * BM;
  int s0 = blockIdx.x * BN;
  const __bf16* theta = TPG + ((size_t)0 * B_ + b) * D_ * T_;  // [d][t]
  const __bf16* phi   = TPG + ((size_t)1 * B_ + b) * D_ * T_;  // [d][s]
  GEMM_PROLOGUE();

  lds_copy_trans(lds_a[0], theta, T_, 0, t0, tid);
  lds_copy_trans(lds_b[0], phi,   T_, 0, s0, tid);
  for (int k0 = 0; k0 < D_; k0 += 2 * BK) {
    __syncthreads();
    lds_copy_trans(lds_a[1], theta, T_, k0 + BK, t0, tid);
    lds_copy_trans(lds_b[1], phi,   T_, k0 + BK, s0, tid);
    if (k0 + 2 * BK < D_) {
      pf_trans(theta, T_, k0 + 2 * BK, t0, tid);
      pf_trans(phi,   T_, k0 + 2 * BK, s0, tid);
    }
    GEMM_MAC(lds_a[0], lds_b[0]);
    __syncthreads();
    if (k0 + 2 * BK < D_) {
      lds_copy_trans(lds_a[0], theta, T_, k0 + 2 * BK, t0, tid);
      lds_copy_trans(lds_b[0], phi,   T_, k0 + 2 * BK, s0, tid);
      pf_trans(theta, T_, k0 + 3 * BK, t0, tid);
      pf_trans(phi,   T_, k0 + 3 * BK, s0, tid);
    }
    GEMM_MAC(lds_a[1], lds_b[1]);
  }

  float* Sb = S + (size_t)b * T_ * T_;
  #pragma unroll
  for (int i = 0; i < 2; ++i)
    #pragma unroll
    for (int j = 0; j < 4; ++j)
      #pragma unroll
      for (int e = 0; e < 8; ++e) {
        int t = t0 + wm + i * 16 + e + 8 * half;
        int s = s0 + wn + j * 16 + col;
        Sb[(size_t)t * T_ + s] = acc[i][j][e];
      }
}

// ---------------------------------------------------------------------------
// Kernel 3: per-row softmax stats: m[t] = max_s S, r[t] = 1/sum_s exp(S-m)
// One wave per row (8 rows per 256-thread block).
// ---------------------------------------------------------------------------
__global__ __launch_bounds__(256)
void k_rowstats(const float* __restrict__ S, float* __restrict__ Mrow,
                float* __restrict__ Rrow) {
  int row  = blockIdx.x * 8 + (threadIdx.x >> 5);   // 0 .. B*T-1
  int lane = threadIdx.x & 31;
  const float* p = S + (size_t)row * T_;
  float m = -1e30f;
  for (int i = lane; i < T_; i += 32) m = fmaxf(m, p[i]);
  #pragma unroll
  for (int d = 16; d >= 1; d >>= 1) m = fmaxf(m, __shfl_xor(m, d, 32));
  float z = 0.f;
  for (int i = lane; i < T_; i += 32) z += __expf(p[i] - m);
  #pragma unroll
  for (int d = 16; d >= 1; d >>= 1) z += __shfl_xor(z, d, 32);
  if (lane == 0) { Mrow[row] = m; Rrow[row] = 1.f / z; }
}

// ---------------------------------------------------------------------------
// Kernel 4: fold 1/Z into g:  GS[b][d][t] = g[b][d][t] * Rrow[b][t]
// ---------------------------------------------------------------------------
__global__ __launch_bounds__(256)
void k_gscale(const __bf16* __restrict__ G, const float* __restrict__ Rrow,
              __bf16* __restrict__ GS) {
  int i0 = (blockIdx.x * 256 + threadIdx.x) * 4;
  #pragma unroll
  for (int j = 0; j < 4; ++j) {
    int i = i0 + j;
    int b = i / (D_ * T_);
    int t = i & (T_ - 1);
    GS[i] = (__bf16)((float)G[i] * Rrow[b * T_ + t]);
  }
}

// ---------------------------------------------------------------------------
// Kernel 5: attn[d][s] = sum_t GS[d][t] * exp(S[t][s] - m[t])   (per batch)
// E tile converted f32->exp->bf16 on the fly into LDS.
// ---------------------------------------------------------------------------
__global__ __launch_bounds__(256)
void k_attn(const __bf16* __restrict__ GS, const float* __restrict__ S,
            const float* __restrict__ Mrow, __bf16* __restrict__ ATT) {
  __shared__ __align__(16) __bf16 lds_a[2][BM * BK];
  __shared__ __align__(16) __bf16 lds_b[2][BN * BK];
  int b  = blockIdx.z;
  int m0 = blockIdx.y * BM;   // over D = 256
  int n0 = blockIdx.x * BN;   // over T (s)
  const __bf16* A  = GS + (size_t)b * D_ * T_;    // [d][t], K = t contiguous
  const float*  Sb = S + (size_t)b * T_ * T_;
  const float*  Mb = Mrow + (size_t)b * T_;
  GEMM_PROLOGUE();

  lds_copy_direct(lds_a[0], A, T_, m0, 0, tid);
  lds_copy_exp   (lds_b[0], Sb, Mb, 0, n0, tid);
  for (int k0 = 0; k0 < T_; k0 += 2 * BK) {
    __syncthreads();
    lds_copy_direct(lds_a[1], A, T_, m0, k0 + BK, tid);
    lds_copy_exp   (lds_b[1], Sb, Mb, k0 + BK, n0, tid);
    if (k0 + 2 * BK < T_) {
      pf_direct(A, T_, m0, k0 + 2 * BK, tid);
      pf_exp   (Sb, k0 + 2 * BK, n0, tid);
    }
    GEMM_MAC(lds_a[0], lds_b[0]);
    __syncthreads();
    if (k0 + 2 * BK < T_) {
      lds_copy_direct(lds_a[0], A, T_, m0, k0 + 2 * BK, tid);
      lds_copy_exp   (lds_b[0], Sb, Mb, k0 + 2 * BK, n0, tid);
      pf_direct(A, T_, m0, k0 + 3 * BK, tid);
      pf_exp   (Sb, k0 + 3 * BK, n0, tid);
    }
    GEMM_MAC(lds_a[1], lds_b[1]);
  }

  __bf16* Ob = ATT + (size_t)b * D_ * T_;
  #pragma unroll
  for (int i = 0; i < 2; ++i)
    #pragma unroll
    for (int j = 0; j < 4; ++j)
      #pragma unroll
      for (int e = 0; e < 8; ++e) {
        int d = m0 + wm + i * 16 + e + 8 * half;
        int s = n0 + wn + j * 16 + col;
        Ob[(size_t)d * T_ + s] = (__bf16)acc[i][j][e];
      }
}

// ---------------------------------------------------------------------------
// Kernel 6: h[c][t] = ht_w[c][d] * attn[d][t] + ht_b[c]; accumulate BN stats.
// ---------------------------------------------------------------------------
__global__ __launch_bounds__(256)
void k_ht(const __bf16* __restrict__ Wht, const __bf16* __restrict__ ATT,
          const float* __restrict__ htb, float* __restrict__ H,
          float* __restrict__ CH) {
  __shared__ __align__(16) __bf16 lds_a[2][BM * BK];
  __shared__ __align__(16) __bf16 lds_b[2][BN * BK];
  int b  = blockIdx.z;
  int m0 = blockIdx.y * BM;   // over C = 512
  int n0 = blockIdx.x * BN;   // over T
  const __bf16* Bt = ATT + (size_t)b * D_ * T_;   // [d][t]
  GEMM_PROLOGUE();

  lds_copy_direct(lds_a[0], Wht, D_, m0, 0, tid);
  lds_copy_trans (lds_b[0], Bt,  T_, 0, n0, tid);
  for (int k0 = 0; k0 < D_; k0 += 2 * BK) {
    __syncthreads();
    lds_copy_direct(lds_a[1], Wht, D_, m0, k0 + BK, tid);
    lds_copy_trans (lds_b[1], Bt,  T_, k0 + BK, n0, tid);
    if (k0 + 2 * BK < D_) {
      pf_direct(Wht, D_, m0, k0 + 2 * BK, tid);
      pf_trans (Bt,  T_, k0 + 2 * BK, n0, tid);
    }
    GEMM_MAC(lds_a[0], lds_b[0]);
    __syncthreads();
    if (k0 + 2 * BK < D_) {
      lds_copy_direct(lds_a[0], Wht, D_, m0, k0 + 2 * BK, tid);
      lds_copy_trans (lds_b[0], Bt,  T_, k0 + 2 * BK, n0, tid);
      pf_direct(Wht, D_, m0, k0 + 3 * BK, tid);
      pf_trans (Bt,  T_, k0 + 3 * BK, n0, tid);
    }
    GEMM_MAC(lds_a[1], lds_b[1]);
  }

  float* Hb = H + (size_t)b * C_ * T_;
  #pragma unroll
  for (int i = 0; i < 2; ++i)
    #pragma unroll
    for (int j = 0; j < 4; ++j)
      #pragma unroll
      for (int e = 0; e < 8; ++e) {
        int c = m0 + wm + i * 16 + e + 8 * half;
        int t = n0 + wn + j * 16 + col;
        float h = acc[i][j][e] + htb[c];
        Hb[(size_t)c * T_ + t] = h;
        // reduce across the 16 lanes sharing this channel row, then one atomic
        float s1 = h, s2 = h * h;
        #pragma unroll
        for (int d2 = 8; d2 >= 1; d2 >>= 1) {
          s1 += __shfl_xor(s1, d2, 32);
          s2 += __shfl_xor(s2, d2, 32);
        }
        if (col == 0) {
          atomicAdd(&CH[c], s1);
          atomicAdd(&CH[C_ + c], s2);
        }
      }
}

// ---------------------------------------------------------------------------
// Kernel 7: out = inpt + gamma*(H - mean)*rsqrt(var+eps) + beta
// ---------------------------------------------------------------------------
__global__ __launch_bounds__(256)
void k_final(const float* __restrict__ inpt, const float* __restrict__ H,
             const float* __restrict__ CH, const float* __restrict__ gamma,
             const float* __restrict__ beta, float* __restrict__ out) {
  int i0 = (blockIdx.x * 256 + threadIdx.x) * 4;
  const float inv = 1.0f / (float)(B_ * T_);
  #pragma unroll
  for (int j = 0; j < 4; ++j) {
    int i = i0 + j;
    int c = (i / T_) & (C_ - 1);
    float mean = CH[c] * inv;
    float var  = CH[C_ + c] * inv - mean * mean;
    out[i] = inpt[i] + gamma[c] * (H[i] - mean) * rsqrtf(var + EPS_) + beta[c];
  }
}

// ---------------------------------------------------------------------------
// Utility kernels
// ---------------------------------------------------------------------------
__global__ __launch_bounds__(256)
void k_cvt(const float* __restrict__ src, __bf16* __restrict__ dst, int n) {
  int i0 = (blockIdx.x * 256 + threadIdx.x) * 4;
  if (i0 + 3 < n) {
    #pragma unroll
    for (int j = 0; j < 4; ++j) dst[i0 + j] = (__bf16)src[i0 + j];
  }
}

__global__ __launch_bounds__(256)
void k_bias_cat(const float* __restrict__ a, const float* __restrict__ b,
                const float* __restrict__ c, float* __restrict__ dst) {
  int i = blockIdx.x * 256 + threadIdx.x;
  if (i < D_)          dst[i] = a[i];
  else if (i < 2 * D_) dst[i] = b[i - D_];
  else if (i < 3 * D_) dst[i] = c[i - 2 * D_];
}

__global__ __launch_bounds__(256)
void k_zero(float* __restrict__ p, int n) {
  int i = blockIdx.x * 256 + threadIdx.x;
  if (i < n) p[i] = 0.f;
}

// ---------------------------------------------------------------------------
// Host launcher
// ---------------------------------------------------------------------------
extern "C" void kernel_launch(void* const* d_in, const int* in_sizes, int n_in,
                              void* d_out, int out_size, void* d_ws, size_t ws_size,
                              hipStream_t stream) {
  const float* inpt  = (const float*)d_in[0];
  const float* thw   = (const float*)d_in[1];
  const float* thb   = (const float*)d_in[2];
  const float* phw   = (const float*)d_in[3];
  const float* phb   = (const float*)d_in[4];
  const float* gw    = (const float*)d_in[5];
  const float* gbv   = (const float*)d_in[6];
  const float* htw   = (const float*)d_in[7];
  const float* htb   = (const float*)d_in[8];
  const float* gamma = (const float*)d_in[9];
  const float* beta  = (const float*)d_in[10];
  float* out = (float*)d_out;

  char* ws = (char*)d_ws;
  size_t off = 0;
  auto take = [&](size_t bytes) -> char* {
    char* p = ws + off;
    off += (bytes + 255) & ~(size_t)255;
    return p;
  };
  __bf16* Xbf  = (__bf16*)take((size_t)B_ * C_ * T_ * 2);     // input, bf16
  __bf16* Wcat = (__bf16*)take((size_t)3 * D_ * C_ * 2);      // theta|phi|g weights
  __bf16* Wht  = (__bf16*)take((size_t)C_ * D_ * 2);          // ht weights
  float*  Bcat = (float*) take((size_t)3 * D_ * 4);           // concat biases
  __bf16* TPG  = (__bf16*)take((size_t)3 * B_ * D_ * T_ * 2); // theta/phi/g acts
  __bf16* GS   = (__bf16*)take((size_t)B_ * D_ * T_ * 2);     // g / Z
  float*  S    = (float*) take((size_t)B_ * T_ * T_ * 4);     // scores
  float*  Mrow = (float*) take((size_t)B_ * T_ * 4);
  float*  Rrow = (float*) take((size_t)B_ * T_ * 4);
  __bf16* ATT  = (__bf16*)take((size_t)B_ * D_ * T_ * 2);     // attn output
  float*  H    = (float*) take((size_t)B_ * C_ * T_ * 4);     // pre-BN
  float*  CH   = (float*) take((size_t)2 * C_ * 4);           // sum, sumsq

  // precision conversion + stat reset
  k_cvt <<<(B_*C_*T_) / 1024, 256, 0, stream>>>(inpt, Xbf, B_*C_*T_);
  k_cvt <<<(D_*C_) / 1024,    256, 0, stream>>>(thw, Wcat,            D_*C_);
  k_cvt <<<(D_*C_) / 1024,    256, 0, stream>>>(phw, Wcat + D_*C_,    D_*C_);
  k_cvt <<<(D_*C_) / 1024,    256, 0, stream>>>(gw,  Wcat + 2*D_*C_,  D_*C_);
  k_cvt <<<(C_*D_) / 1024,    256, 0, stream>>>(htw, Wht,             C_*D_);
  k_bias_cat<<<3, 256, 0, stream>>>(thb, phb, gbv, Bcat);
  k_zero<<<(2*C_ + 255) / 256, 256, 0, stream>>>(CH, 2*C_);

  // pipeline
  k_proj    <<<dim3(T_/BN, (3*D_)/BM, B_), 256, 0, stream>>>(Wcat, Xbf, Bcat, TPG);
  k_scores  <<<dim3(T_/BN, T_/BM, B_),     256, 0, stream>>>(TPG, S);
  k_rowstats<<<(B_*T_)/8,                  256, 0, stream>>>(S, Mrow, Rrow);
  k_gscale  <<<(B_*D_*T_)/1024,            256, 0, stream>>>(TPG + (size_t)2*B_*D_*T_, Rrow, GS);
  k_attn    <<<dim3(T_/BN, D_/BM, B_),     256, 0, stream>>>(GS, S, Mrow, ATT);
  k_ht      <<<dim3(T_/BN, C_/BM, B_),     256, 0, stream>>>(Wht, ATT, htb, H, CH);
  k_final   <<<(B_*C_*T_)/1024,            256, 0, stream>>>(inpt, H, CH, gamma, beta, out);
}
